// LTLoss_59966333387173
// MI455X (gfx1250) — matrix operand order repaired and checked by reading, not compiled, not run on hardware
//
#include <hip/hip_runtime.h>
#include <hip/hip_bf16.h>
#include <math.h>

// ---------------------------------------------------------------------------
// Problem constants (from reference): B=16384, C=1000, D=512
// Combined GEMM: M=B, N=Cpad=1024 (zero-padded), K=2D=1024 in bf16 WMMA.
// ---------------------------------------------------------------------------
#define B_N    16384
#define C_N    1000
#define D_N    512
#define CPAD   1024
#define KDIM   1024          // 2*D
#define KT_N   32            // KDIM / 32  (k-steps of the 16x16x32 WMMA)
#define NT_N   64            // CPAD / 16  (N tiles)
#define LOG_D  6.238324625039508f   // ln(512)
#define ALPHA_C  0.1f
#define LAMBDA_C 0.01f

typedef __bf16  bf16_t;
typedef bf16_t  v16bf __attribute__((ext_vector_type(16)));
typedef float   v8f   __attribute__((ext_vector_type(8)));

// ---------------- reduction helpers (wave32, 256-thread blocks) ------------
__device__ inline float block_reduce_sum(float v, float* sm) {
  int lane = threadIdx.x & 31;
  int wid  = threadIdx.x >> 5;
#pragma unroll
  for (int o = 16; o > 0; o >>= 1) v += __shfl_xor(v, o, 32);
  if (lane == 0) sm[wid] = v;
  __syncthreads();
  if (threadIdx.x == 0) {
    float r = 0.f;
#pragma unroll
    for (int i = 0; i < 8; ++i) r += sm[i];
    sm[0] = r;
  }
  __syncthreads();
  float r = sm[0];
  __syncthreads();
  return r;
}

__device__ inline float block_reduce_prod(float v, float* sm) {
  int lane = threadIdx.x & 31;
  int wid  = threadIdx.x >> 5;
#pragma unroll
  for (int o = 16; o > 0; o >>= 1) v *= __shfl_xor(v, o, 32);
  if (lane == 0) sm[wid] = v;
  __syncthreads();
  if (threadIdx.x == 0) {
    float r = 1.f;
#pragma unroll
    for (int i = 0; i < 8; ++i) r *= sm[i];
    sm[0] = r;
  }
  __syncthreads();
  float r = sm[0];
  __syncthreads();
  return r;
}

// ---------------------------------------------------------------------------
// Kernel 1: per-class prep. Normalize means, per-class constants, and pack
// one column of the combined B matrix [w ; -2*mhat*w] into WMMA B-fragment
// layout: lane l holds N = l&15; bf16 element j holds K = (j&7) + (l>=16 ? 8:0) + (j>=8 ? 16:0).
// 256 threads, one block per (padded) class.
// ---------------------------------------------------------------------------
__global__ __launch_bounds__(256)
void prep_classes(const float* __restrict__ means, const float* __restrict__ var,
                  const float* __restrict__ rawfd,
                  float* __restrict__ mhat, bf16_t* __restrict__ packedB,
                  float* __restrict__ fdArr, float* __restrict__ exArr,
                  float* __restrict__ coefArr, float* __restrict__ sArr) {
  __shared__ float sm[8];
  __shared__ float sh_inv;
  const int c = blockIdx.x;
  const int t = threadIdx.x;
  const bool valid = (c < C_N);

  float m0 = 0.f, m1 = 0.f, v0 = 1.f, v1 = 1.f;
  if (valid) {
    m0 = means[(size_t)c * D_N + t];
    m1 = means[(size_t)c * D_N + t + 256];
    v0 = var  [(size_t)c * D_N + t];
    v1 = var  [(size_t)c * D_N + t + 256];
  }
  float ss = block_reduce_sum(m0 * m0 + m1 * m1, sm);
  if (t == 0) sh_inv = 1.f / fmaxf(sqrtf(ss), 1e-12f);
  __syncthreads();
  const float inv = sh_inv;

  const float mh0 = m0 * inv, mh1 = m1 * inv;
  const float w0 = 1.f / (v0 + 1e-8f), w1 = 1.f / (v1 + 1e-8f);
  if (valid) {
    mhat[(size_t)c * D_N + t]       = mh0;
    mhat[(size_t)c * D_N + t + 256] = mh1;
  }
  float s   = block_reduce_sum (mh0 * mh0 * w0 + mh1 * mh1 * w1, sm);
  float det = block_reduce_prod(v0 * v1, sm);

  if (t == 0) {
    if (valid) {
      float r  = rawfd[c];
      float fd = r * r + 0.1f;
      float ex = 0.5f * (fd + LOG_D);
      float cup = expf(lgammaf(fminf(fmaxf(ex, 1e-8f), 1e8f)));
      float cdn = expf(lgammaf(fminf(fmaxf(0.5f * fd, 1e-8f), 1e8f))) *
                  powf(fd, 0.5f * LOG_D);
      fdArr[c]   = fd;
      exArr[c]   = ex;
      coefArr[c] = cup / cdn * sqrtf(det);
      sArr[c]    = s;
    } else {
      fdArr[c] = 1.f; exArr[c] = 1.f; coefArr[c] = 0.f; sArr[c] = 0.f;
    }
  }

  // Pack column c of the combined 1024 x CPAD B matrix.
  const int nt = c >> 4, ln = c & 15;
  for (int k = t; k < KDIM; k += 256) {
    float val = 0.f;
    if (valid) {
      int d   = k & (D_N - 1);
      float w = 1.f / (var[(size_t)c * D_N + d] + 1e-8f);
      val = (k < D_N) ? w : (-2.f * means[(size_t)c * D_N + d] * inv * w);
    }
    int kt = k >> 5, r = k & 31;
    int hi = (r >> 3) & 1, j = (r & 7) + ((r >> 4) << 3);
    int lanep = ln + (hi << 4);
    packedB[(size_t)(kt * NT_N + nt) * 512 + lanep * 16 + j] = (bf16_t)val;
  }
}

// ---------------------------------------------------------------------------
// Kernel 2: per-row feat prep. Normalize, pack A = [fhat^2 | fhat] into WMMA
// A-fragment layout (ISA 16-bit A 16x32): lane<16: j<8->K=j, j>=8->K=j+8;
// lane>=16: +8. One block (256 threads) per row.
// ---------------------------------------------------------------------------
__global__ __launch_bounds__(256)
void prep_feat(const float* __restrict__ feat, bf16_t* __restrict__ packedA) {
  __shared__ float sm[8];
  __shared__ float sh_inv;
  const int b = blockIdx.x;
  const int t = threadIdx.x;
  float f0 = feat[(size_t)b * D_N + t];
  float f1 = feat[(size_t)b * D_N + t + 256];
  float ss = block_reduce_sum(f0 * f0 + f1 * f1, sm);
  if (t == 0) sh_inv = 1.f / fmaxf(sqrtf(ss), 1e-12f);
  __syncthreads();
  const float inv = sh_inv;

  const int mt = b >> 4, m = b & 15;
  for (int k = t; k < KDIM; k += 256) {
    int d = k & (D_N - 1);
    float fh  = feat[(size_t)b * D_N + d] * inv;
    float val = (k < D_N) ? fh * fh : fh;
    int kt = k >> 5, r = k & 31;
    int hi, j;
    if      (r < 8)  { hi = 0; j = r;      }
    else if (r < 16) { hi = 1; j = r - 8;  }
    else if (r < 24) { hi = 0; j = r - 8;  }
    else             { hi = 1; j = r - 16; }
    int lanep = m + (hi << 4);
    packedA[(size_t)(mt * KT_N + kt) * 512 + lanep * 16 + j] = (bf16_t)val;
  }
}

// ---------------------------------------------------------------------------
// Kernel 3: pure WMMA GEMM, raw d2 accumulators stored to out.
// Block = 128 threads = 4 waves (2x2); block tile 128(M) x 128(N); each wave
// owns a 64x64 patch = 4x4 accumulators. The transcendental epilogue lives in
// logit_finish (which re-reads out anyway), so post-loop register demand here
// is tiny and the ~194-VGPR loop working set does not spill.
// ---------------------------------------------------------------------------
__global__ __launch_bounds__(128)
void gemm_raw(const bf16_t* __restrict__ packedA, const bf16_t* __restrict__ packedB,
              float* __restrict__ out) {
  const int tid  = threadIdx.x;
  const int lane = tid & 31;
  const int wave = tid >> 5;         // 0..3
  const int wm = wave >> 1;          // 0..1
  const int wn = wave & 1;           // 0..1
  const int mt0 = blockIdx.x * 8 + wm * 4;   // 8 M tiles (16 rows) per block
  const int nt0 = blockIdx.y * 8 + wn * 4;   // 8 N tiles (16 cols) per block

  v8f acc[4][4] = {};
  const bf16_t* aBase[4];
  const bf16_t* bBase[4];
#pragma unroll
  for (int i = 0; i < 4; ++i)
    aBase[i] = packedA + (size_t)(mt0 + i) * KT_N * 512 + lane * 16;
#pragma unroll
  for (int j = 0; j < 4; ++j)
    bBase[j] = packedB + (size_t)(nt0 + j) * 512 + lane * 16;

  for (int kt = 0; kt < KT_N; ++kt) {
    v16bf a[4], b[4];
#pragma unroll
    for (int i = 0; i < 4; ++i) a[i] = *(const v16bf*)(aBase[i] + (size_t)kt * 512);
#pragma unroll
    for (int j = 0; j < 4; ++j) b[j] = *(const v16bf*)(bBase[j] + (size_t)kt * NT_N * 512);
    if (kt + 1 < KT_N) {   // pull the strided B stream toward the near caches
#pragma unroll
      for (int j = 0; j < 4; ++j)
        __builtin_prefetch(bBase[j] + (size_t)(kt + 1) * NT_N * 512, 0, 3);
    }
#pragma unroll
    for (int i = 0; i < 4; ++i)
#pragma unroll
      for (int j = 0; j < 4; ++j)
        acc[i][j] = __builtin_amdgcn_wmma_f32_16x16x32_bf16(
            false, a[i], false, b[j], (short)0, acc[i][j], false, false);
  }

  // C/D layout: VGPR v, lanes 0-15 -> (M=v, N=lane); lanes 16-31 -> (M=8+v, N=lane-16)
  const int colInLane = lane & 15;
  const int rowOff    = (lane < 16) ? 0 : 8;
#pragma unroll
  for (int j = 0; j < 4; ++j) {
    int c = (nt0 + j) * 16 + colInLane;
    if (c >= C_N) continue;
#pragma unroll
    for (int i = 0; i < 4; ++i) {
      int bRowBase = (mt0 + i) * 16 + rowOff;
#pragma unroll
      for (int v = 0; v < 8; ++v)
        out[(size_t)(bRowBase + v) * C_N + c] = acc[i][j][v];
    }
  }
}

// ---------------------------------------------------------------------------
// Kernel 4: fused logit epilogue + per-row L2 normalization (in place).
// Each thread handles 4 consecutive classes (one float4 of the row AND one
// float4 of each per-class constant array). Student-t math, row sum-of-
// squares reduction, then rescale — one read + one write of the row.
// ---------------------------------------------------------------------------
__global__ __launch_bounds__(256)
void logit_finish(float* __restrict__ out,
                  const float* __restrict__ fdArr, const float* __restrict__ exArr,
                  const float* __restrict__ coefArr, const float* __restrict__ sArr,
                  const int* __restrict__ labels) {
  __shared__ float sm[8];
  __shared__ float sh_inv;
  const int b = blockIdx.x;
  const int t = threadIdx.x;
  float4* row = (float4*)(out + (size_t)b * C_N);
  const int y = labels[b];

  float4 lv = make_float4(0.f, 0.f, 0.f, 0.f);
  float ss = 0.f;
  if (t < 250) {
    float4 raw  = row[t];
    int    c0   = t * 4;
    float4 fd4  = *(const float4*)(fdArr   + c0);
    float4 ex4  = *(const float4*)(exArr   + c0);
    float4 cf4  = *(const float4*)(coefArr + c0);
    float4 s4   = *(const float4*)(sArr    + c0);
    float k0 = (y == c0 + 0) ? (1.f + ALPHA_C) : 1.f;
    float k1 = (y == c0 + 1) ? (1.f + ALPHA_C) : 1.f;
    float k2 = (y == c0 + 2) ? (1.f + ALPHA_C) : 1.f;
    float k3 = (y == c0 + 3) ? (1.f + ALPHA_C) : 1.f;
    lv.x = cf4.x * powf(1.f + k0 * 0.5f * (raw.x + s4.x) / fd4.x + 1e-8f, -ex4.x);
    lv.y = cf4.y * powf(1.f + k1 * 0.5f * (raw.y + s4.y) / fd4.y + 1e-8f, -ex4.y);
    lv.z = cf4.z * powf(1.f + k2 * 0.5f * (raw.z + s4.z) / fd4.z + 1e-8f, -ex4.z);
    lv.w = cf4.w * powf(1.f + k3 * 0.5f * (raw.w + s4.w) / fd4.w + 1e-8f, -ex4.w);
    ss = lv.x * lv.x + lv.y * lv.y + lv.z * lv.z + lv.w * lv.w;
  }
  ss = block_reduce_sum(ss, sm);
  if (t == 0) sh_inv = 1.f / fmaxf(sqrtf(ss), 1e-12f);
  __syncthreads();
  const float inv = sh_inv;
  if (t < 250) {
    lv.x *= inv; lv.y *= inv; lv.z *= inv; lv.w *= inv;
    row[t] = lv;
  }
}

// ---------------------------------------------------------------------------
// Kernel 5/6: center loss. Both fhat and mhat are unit vectors, so
// ||fhat - mhat_y||^2 = 2 - 2*fhat.mhat_y. Wave-per-row dot products.
// ---------------------------------------------------------------------------
__global__ __launch_bounds__(256)
void center_partial(const float* __restrict__ feat, const float* __restrict__ mhat,
                    const int* __restrict__ labels, float* __restrict__ partials) {
  __shared__ float sm[8];
  const int wave = threadIdx.x >> 5, lane = threadIdx.x & 31;
  float accum = 0.f;
  const int rowBase = blockIdx.x * 64 + wave * 8;
#pragma unroll 1
  for (int rr = 0; rr < 8; ++rr) {
    int r = rowBase + rr;
    int y = labels[r];
    const float* fr = feat + (size_t)r * D_N;
    const float* mr = mhat + (size_t)y * D_N;
    float sn = 0.f, sd = 0.f;
#pragma unroll
    for (int i = 0; i < 16; ++i) {
      float f  = fr[i * 32 + lane];
      float mh = mr[i * 32 + lane];
      sn += f * f;
      sd += f * mh;
    }
#pragma unroll
    for (int o = 16; o > 0; o >>= 1) {
      sn += __shfl_xor(sn, o, 32);
      sd += __shfl_xor(sd, o, 32);
    }
    if (lane == 0) accum += 1.f - sd / fmaxf(sqrtf(sn), 1e-12f);
  }
  if (lane == 0) sm[wave] = accum;
  __syncthreads();
  if (threadIdx.x == 0) {
    float tot = 0.f;
#pragma unroll
    for (int i = 0; i < 8; ++i) tot += sm[i];
    partials[blockIdx.x] = tot;
  }
}

__global__ __launch_bounds__(256)
void center_final(const float* __restrict__ partials, float* __restrict__ out) {
  __shared__ float sm[8];
  float tot = block_reduce_sum(partials[threadIdx.x], sm);
  if (threadIdx.x == 0) out[(size_t)B_N * C_N] = (LAMBDA_C / (float)B_N) * tot;
}

// ---------------------------------------------------------------------------
extern "C" void kernel_launch(void* const* d_in, const int* in_sizes, int n_in,
                              void* d_out, int out_size, void* d_ws, size_t ws_size,
                              hipStream_t stream) {
  (void)in_sizes; (void)n_in; (void)out_size; (void)ws_size;
  const float* feat   = (const float*)d_in[0];   // [B, D]
  const float* means  = (const float*)d_in[1];   // [C, D]
  const float* rawfd  = (const float*)d_in[2];   // [C, 1]
  const float* var    = (const float*)d_in[3];   // [C, D]
  const int*   labels = (const int*)  d_in[4];   // [B]
  float* out = (float*)d_out;                    // [B*C] logits + [1] center loss

  // Workspace layout (~37.7 MB total)
  char* ws = (char*)d_ws;
  size_t off = 0;
  bf16_t* packedA = (bf16_t*)(ws + off); off += (size_t)B_N * KDIM * sizeof(bf16_t);   // 32 MB
  bf16_t* packedB = (bf16_t*)(ws + off); off += (size_t)CPAD * KDIM * sizeof(bf16_t);  // 2 MB
  float*  mhat    = (float*) (ws + off); off += (size_t)C_N * D_N * sizeof(float);     // 2 MB
  float*  fdArr   = (float*) (ws + off); off += CPAD * sizeof(float);
  float*  exArr   = (float*) (ws + off); off += CPAD * sizeof(float);
  float*  coefArr = (float*) (ws + off); off += CPAD * sizeof(float);
  float*  sArr    = (float*) (ws + off); off += CPAD * sizeof(float);
  float*  parts   = (float*) (ws + off); off += 256 * sizeof(float);

  prep_classes<<<CPAD, 256, 0, stream>>>(means, var, rawfd, mhat, packedB,
                                         fdArr, exArr, coefArr, sArr);
  prep_feat<<<B_N, 256, 0, stream>>>(feat, packedA);

  dim3 gGrid(B_N / 128, CPAD / 128);   // (128, 8)
  gemm_raw<<<gGrid, 128, 0, stream>>>(packedA, packedB, out);

  logit_finish<<<B_N, 256, 0, stream>>>(out, fdArr, exArr, coefArr, sArr, labels);

  center_partial<<<B_N / 64, 256, 0, stream>>>(feat, mhat, labels, parts);
  center_final<<<1, 256, 0, stream>>>(parts, out);
}
